// GreedyDecoder_14525579395604
// MI455X (gfx1250) — matrix operand order repaired and checked by reading, not compiled, not run on hardware
//
#include <hip/hip_runtime.h>

// ---------------------------------------------------------------------------
// CDNA5 (gfx1250) WMMA bf16 helpers
// ---------------------------------------------------------------------------
typedef __attribute__((ext_vector_type(16))) __bf16 v16bf;
typedef __attribute__((ext_vector_type(8)))  float  v8f;

union BFrag { v16bf v; uint4 q[2]; };

__device__ __forceinline__ v8f wmma_bf16(v16bf a, v16bf b, v8f c) {
  return __builtin_amdgcn_wmma_f32_16x16x32_bf16(false, a, false, b, (short)0, c,
                                                 false, false);
}

__device__ __forceinline__ unsigned short f2bf(float x) {  // round-to-nearest-even
  unsigned int u = __float_as_uint(x);
  u += 0x7fffu + ((u >> 16) & 1u);
  return (unsigned short)(u >> 16);
}

__device__ __forceinline__ float sigf(float x) { return 1.0f / (1.0f + __expf(-x)); }

// A fragment: 16(M) x 32(K) bf16 tile of a row-major activation matrix.
// ISA layout: lanes 0-15 hold M=lane, K = k0+{0..7,16..23}; lanes 16-31: K = k0+{8..15,24..31}.
// Optional `rows` remaps logical M-row -> absolute source row (used for fi=f[tid,n]).
__device__ __forceinline__ v16bf load_a_frag(const unsigned short* A, int lda,
                                             int m16, int k0, int lane,
                                             const int* rows) {
  int m = m16 + (lane & 15);
  int r = rows ? rows[m] : m;
  const unsigned short* p = A + (size_t)r * lda + k0 + ((lane & 16) ? 8 : 0);
  BFrag f;
  f.q[0] = *(const uint4*)p;
  f.q[1] = *(const uint4*)(p + 16);
  return f.v;
}

// B fragment: 32(K) x 16(N) bf16 tile of W^T where W is row-major [O,K] (ldw=K).
// Lane = N (col of output = row of W); lanes 0-15: K = k0+{0..7,16..23}; lanes 16-31: K = k0+{8..15,24..31}.
__device__ __forceinline__ v16bf load_b_frag(const unsigned short* W, int ldw,
                                             int o16, int k0, int lane) {
  const unsigned short* p =
      W + (size_t)(o16 + (lane & 15)) * ldw + k0 + ((lane & 16) ? 8 : 0);
  BFrag f;
  f.q[0] = *(const uint4*)p;
  f.q[1] = *(const uint4*)(p + 16);
  return f.v;
}

// ---------------------------------------------------------------------------
// fp32 -> bf16 conversion with row/col zero padding and column offset (for
// [Wih|Whh] concatenation).  dst[r*dstLd + colOff + c] for c in [0,padCols).
// ---------------------------------------------------------------------------
__global__ void __launch_bounds__(256) cvt_bf16_kernel(
    unsigned short* __restrict__ dst, int dstLd, int colOff,
    const float* __restrict__ src, int srcRows, int srcCols,
    int padRows, int padCols) {
  long i = (long)blockIdx.x * blockDim.x + threadIdx.x;
  long total = (long)padRows * padCols;
  if (i >= total) return;
  int r = (int)(i / padCols);
  int c = (int)(i - (long)r * padCols);
  float v = (r < srcRows && c < srcCols) ? src[(size_t)r * srcCols + c] : 0.0f;
  dst[(size_t)r * dstLd + colOff + c] = f2bf(v);
}

__global__ void __launch_bounds__(256) zero_state_kernel(
    float* __restrict__ h, float* __restrict__ c, unsigned short* __restrict__ hb,
    int n) {
  int i = blockIdx.x * blockDim.x + threadIdx.x;
  if (i < n) { h[i] = 0.0f; c[i] = 0.0f; hb[i] = 0; }
}

// ---------------------------------------------------------------------------
// One recurrent LSTM step, gates GEMM:
//   Z[64,O] = A1[64,K1] @ W1[O,K1]^T + A2[64,K2] @ W2[O,K2]^T + bias
// A*, W* bf16 (row-major, ld == K), Z fp32.  Each wave owns a 16x64 strip.
// ---------------------------------------------------------------------------
__global__ void __launch_bounds__(256) lstm_gates_kernel(
    const unsigned short* __restrict__ A1, int lda1, int K1,
    const unsigned short* __restrict__ W1,
    const unsigned short* __restrict__ A2, int K2,
    const unsigned short* __restrict__ W2,
    const float* __restrict__ bias, float* __restrict__ Z, int O) {
  int gwave = (int)((blockIdx.x * blockDim.x + threadIdx.x) >> 5);
  int lane = threadIdx.x & 31;
  int nWaves = (int)((gridDim.x * blockDim.x) >> 5);
  int nUnits = (O >> 6) << 2;  // 4 M-tiles x O/64 strips
  for (int u = gwave; u < nUnits; u += nWaves) {
    int m16 = (u & 3) << 4;
    int o64 = (u >> 2) << 6;
    v8f acc0 = {}, acc1 = {}, acc2 = {}, acc3 = {};
    for (int k = 0; k < K1; k += 32) {
      v16bf a = load_a_frag(A1, lda1, m16, k, lane, nullptr);
      acc0 = wmma_bf16(a, load_b_frag(W1, K1, o64,      k, lane), acc0);
      acc1 = wmma_bf16(a, load_b_frag(W1, K1, o64 + 16, k, lane), acc1);
      acc2 = wmma_bf16(a, load_b_frag(W1, K1, o64 + 32, k, lane), acc2);
      acc3 = wmma_bf16(a, load_b_frag(W1, K1, o64 + 48, k, lane), acc3);
    }
    for (int k = 0; k < K2; k += 32) {
      v16bf a = load_a_frag(A2, K2, m16, k, lane, nullptr);
      acc0 = wmma_bf16(a, load_b_frag(W2, K2, o64,      k, lane), acc0);
      acc1 = wmma_bf16(a, load_b_frag(W2, K2, o64 + 16, k, lane), acc1);
      acc2 = wmma_bf16(a, load_b_frag(W2, K2, o64 + 32, k, lane), acc2);
      acc3 = wmma_bf16(a, load_b_frag(W2, K2, o64 + 48, k, lane), acc3);
    }
    int hi8 = (lane & 16) ? 8 : 0;
    int oc = lane & 15;
    float b0 = bias[o64 + oc],      b1 = bias[o64 + 16 + oc];
    float b2 = bias[o64 + 32 + oc], b3 = bias[o64 + 48 + oc];
#pragma unroll
    for (int r = 0; r < 8; ++r) {
      float* zr = Z + (size_t)(m16 + r + hi8) * O;
      zr[o64 + oc]      = acc0[r] + b0;
      zr[o64 + 16 + oc] = acc1[r] + b1;
      zr[o64 + 32 + oc] = acc2[r] + b2;
      zr[o64 + 48 + oc] = acc3[r] + b3;
    }
  }
}

// Pointwise LSTM update for the H=1024 transcription layers (torch gate order).
__global__ void __launch_bounds__(256) lstm_pointwise_kernel(
    const float* __restrict__ Z, float* __restrict__ h, float* __restrict__ c,
    unsigned short* __restrict__ hbuf, unsigned short* __restrict__ seq_out) {
  int i = blockIdx.x * blockDim.x + threadIdx.x;
  if (i >= 64 * 1024) return;
  int n = i >> 10, u = i & 1023;
  const float* z = Z + (size_t)n * 4096;
  float cc = sigf(z[1024 + u]) * c[i] + sigf(z[u]) * tanhf(z[2048 + u]);
  float hh = sigf(z[3072 + u]) * tanhf(cc);
  c[i] = cc; h[i] = hh;
  unsigned short hb = f2bf(hh);
  hbuf[i] = hb;
  seq_out[i] = hb;
}

// stack_time: [128,64,1024] -> [64,64,2048] (concat consecutive frame pairs)
__global__ void __launch_bounds__(256) stack_kernel(
    unsigned short* __restrict__ dst, const unsigned short* __restrict__ src) {
  int i = blockIdx.x * blockDim.x + threadIdx.x;
  if (i >= 64 * 64 * 2048) return;
  int u = i & 1023;
  int s = (i >> 10) & 1;
  int n = (i >> 11) & 63;
  int t2 = i >> 17;
  dst[i] = src[((size_t)(2 * t2 + s) * 64 + n) * 1024 + u];
}

// ---------------------------------------------------------------------------
// Persistent greedy-decode kernel: single workgroup, 32 waves.
// ---------------------------------------------------------------------------
__device__ void block_gemm(float* Z, int ldz, int O,
    const unsigned short* A1, int lda1, int K1, const int* rows1,
    const unsigned short* W1,
    const unsigned short* A2, int lda2, int K2, const unsigned short* W2,
    const float* bias, int wave, int lane, int nwaves) {
  int nUnits = (O >> 4) << 2;  // 4 M-tiles x O/16 tiles
  for (int u = wave; u < nUnits; u += nwaves) {
    int m16 = (u & 3) << 4;
    int o16 = (u >> 2) << 4;
    v8f acc = {};
    for (int k = 0; k < K1; k += 32)
      acc = wmma_bf16(load_a_frag(A1, lda1, m16, k, lane, rows1),
                      load_b_frag(W1, K1, o16, k, lane), acc);
    if (A2)
      for (int k = 0; k < K2; k += 32)
        acc = wmma_bf16(load_a_frag(A2, lda2, m16, k, lane, nullptr),
                        load_b_frag(W2, K2, o16, k, lane), acc);
    int hi8 = (lane & 16) ? 8 : 0;
    int oc = o16 + (lane & 15);
    float bs = bias ? bias[oc] : 0.0f;
#pragma unroll
    for (int r = 0; r < 8; ++r)
      Z[(size_t)(m16 + r + hi8) * ldz + oc] = acc[r] + bs;
  }
}

struct DecodeArgs {
  const unsigned short *fenc, *wpred0, *wpred1, *wWe, *wWp, *wWo, *embb;
  const float *bpred0, *bpred1, *bj, *bo32;
  float *hg, *cg, *hgn, *cgn, *zp, *zj, *logits;
  unsigned short *act, *gout, *hj;
  int *pre_g, *sym_add, *tidv, *finishv, *res_idx, *flen, *eosv, *upd, *fi_rows, *done;
  int *res, *counts;
};

__global__ void __launch_bounds__(1024) decode_kernel(DecodeArgs a) {
  const int t = threadIdx.x;
  const int wave = t >> 5, lane = t & 31;
  const int NW = 32;
  const int MAXIT = 64 * 31;  // t_enc * (MAX_SYM+1)
  for (int it = 0; it < MAXIT; ++it) {
    if (a.done[0]) break;
    // --- pred layer 0 input: [emb(pre_g) | bf16(hg0)] ---
    for (int i = t; i < 64 * 320; i += 1024) {
      int n = i / 320, u = i - n * 320;
      int g = a.pre_g[n];
      a.act[n * 640 + u] = (g < 0) ? (unsigned short)0 : a.embb[g * 320 + u];
      a.act[n * 640 + 320 + u] = f2bf(a.hg[i]);
    }
    __threadfence_block(); __syncthreads();
    block_gemm(a.zp, 1280, 1280, a.act, 640, 640, nullptr, a.wpred0,
               nullptr, 0, 0, nullptr, a.bpred0, wave, lane, NW);
    __threadfence_block(); __syncthreads();
    // --- pred layer 0 pointwise; build layer 1 input ---
    for (int i = t; i < 64 * 320; i += 1024) {
      int n = i / 320, u = i - n * 320;
      const float* z = a.zp + (size_t)n * 1280;
      float cc = sigf(z[320 + u]) * a.cg[i] + sigf(z[u]) * tanhf(z[640 + u]);
      float hh = sigf(z[960 + u]) * tanhf(cc);
      a.cgn[i] = cc; a.hgn[i] = hh;
      a.act[n * 640 + u] = f2bf(hh);
      a.act[n * 640 + 320 + u] = f2bf(a.hg[64 * 320 + i]);
    }
    __threadfence_block(); __syncthreads();
    block_gemm(a.zp, 1280, 1280, a.act, 640, 640, nullptr, a.wpred1,
               nullptr, 0, 0, nullptr, a.bpred1, wave, lane, NW);
    __threadfence_block(); __syncthreads();
    // --- pred layer 1 pointwise -> g_out ---
    for (int i = t; i < 64 * 320; i += 1024) {
      int n = i / 320, u = i - n * 320;
      const float* z = a.zp + (size_t)n * 1280;
      int j = 64 * 320 + i;
      float cc = sigf(z[320 + u]) * a.cg[j] + sigf(z[u]) * tanhf(z[640 + u]);
      float hh = sigf(z[960 + u]) * tanhf(cc);
      a.cgn[j] = cc; a.hgn[j] = hh;
      a.gout[i] = f2bf(hh);
    }
    __threadfence_block(); __syncthreads();
    // --- joint hidden: zj = fi@We^T + g@Wp^T + (be+bp); fi gathered via fi_rows ---
    block_gemm(a.zj, 512, 512, a.fenc, 1024, 1024, a.fi_rows, a.wWe,
               a.gout, 320, 320, a.wWp, a.bj, wave, lane, NW);
    __threadfence_block(); __syncthreads();
    for (int i = t; i < 64 * 512; i += 1024) a.hj[i] = f2bf(fmaxf(a.zj[i], 0.0f));
    __threadfence_block(); __syncthreads();
    // --- logits (O padded 29->32; pad bias = -1e30) ---
    block_gemm(a.logits, 32, 32, a.hj, 512, 512, nullptr, a.wWo,
               nullptr, 0, 0, nullptr, a.bo32, wave, lane, NW);
    __threadfence_block(); __syncthreads();
    // --- control: argmax + greedy state update (reference semantics) ---
    if (t < 64) {
      int n = t;
      const float* lg = a.logits + n * 32;
      int best = 0; float bv = lg[0];
      for (int j = 1; j < 29; ++j) { float v = lg[j]; if (v > bv) { bv = v; best = j; } }
      int fin = a.finishv[n];
      int upd_g = (best != 28) && (a.sym_add[n] != 30) && !fin;
      if (upd_g) {
        int ri = a.res_idx[n] + 1;
        a.res_idx[n] = ri;
        a.res[n * 3840 + ri] = best;
        a.sym_add[n] += 1;
        a.pre_g[n] = best;
      } else if (!fin) {
        int tt = a.tidv[n] + 1;
        if (tt >= a.flen[n]) a.finishv[n] = 1;
        tt = min(tt, a.eosv[n]);
        a.tidv[n] = tt;
        a.sym_add[n] = 0;
      }
      a.upd[n] = upd_g;
      a.fi_rows[n] = a.tidv[n] * 64 + n;
    }
    __threadfence_block(); __syncthreads();
    // --- commit pred state where a symbol was emitted ---
    for (int i = t; i < 2 * 64 * 320; i += 1024) {
      int n = (i / 320) & 63;
      if (a.upd[n]) { a.hg[i] = a.hgn[i]; a.cg[i] = a.cgn[i]; }
    }
    if (t == 0) {
      int all = 1;
      for (int n = 0; n < 64; ++n) all &= a.finishv[n];
      a.done[0] = all;
    }
    __threadfence_block(); __syncthreads();
  }
  for (int n = t; n < 64; n += 1024) a.counts[n] = a.res_idx[n] + 1;
}

__global__ void __launch_bounds__(256) decode_init_kernel(
    int* res, int* pre_g, int* sym_add, int* tidv, int* finishv, int* res_idx,
    int* flen, int* eosv, int* upd, int* fi_rows, int* done,
    float* hg, float* cg, float* bj, const float* be, const float* bp,
    float* bo32, const float* bo, const int* x_lens) {
  int i = blockIdx.x * blockDim.x + threadIdx.x;
  if (i < 64 * 3840) res[i] = -1;
  if (i < 2 * 64 * 320) { hg[i] = 0.0f; cg[i] = 0.0f; }
  if (i < 512) bj[i] = be[i] + bp[i];
  if (i < 32) bo32[i] = (i < 29) ? bo[i] : -1.0e30f;
  if (i < 64) {
    int xl = x_lens[i];
    int fl = (xl + 1) >> 1;          // ceil(x_lens/2)
    flen[i] = fl;
    eosv[i] = max(fl - 1, 0);
    pre_g[i] = -1; sym_add[i] = 0; tidv[i] = 0; res_idx[i] = -1;
    finishv[i] = (xl == 0) ? 1 : 0;
    fi_rows[i] = i;                  // fi0 = f[0]
    upd[i] = 0;
  }
  if (i == 0) {
    int all = 1;
    for (int n = 0; n < 64; ++n) all &= (x_lens[n] == 0) ? 1 : 0;
    done[0] = all;
  }
}

// ---------------------------------------------------------------------------
// Host side
// ---------------------------------------------------------------------------
#define CDIV(a, b) (((a) + (b)-1) / (b))

extern "C" void kernel_launch(void* const* d_in, const int* in_sizes, int n_in,
                              void* d_out, int out_size, void* d_ws, size_t ws_size,
                              hipStream_t stream) {
  (void)out_size; (void)ws_size; (void)n_in;
  // ---- resolve input indices (handles insertion-order and sorted-pytree) ----
  int xI, xlI;
  int We_i, Wo_i, Wp_i, be_i, bo_i, bp_i, emb_i;
  int wih_i[5], whh_i[5], b_i[5];          // pre0, pre1, post0, post1, post2
  int pwih_i[2], pwhh_i[2], pb_i[2];
  auto schemeA = [&](int b0) {  // sorted pytree under params
    We_i = b0; Wo_i = b0 + 1; Wp_i = b0 + 2; be_i = b0 + 3; bo_i = b0 + 4;
    bp_i = b0 + 5; emb_i = b0 + 6;
    whh_i[2] = b0 + 7;  wih_i[2] = b0 + 8;  b_i[2] = b0 + 9;    // post0
    whh_i[3] = b0 + 10; wih_i[3] = b0 + 11; b_i[3] = b0 + 12;   // post1
    whh_i[4] = b0 + 13; wih_i[4] = b0 + 14; b_i[4] = b0 + 15;   // post2
    whh_i[0] = b0 + 16; wih_i[0] = b0 + 17; b_i[0] = b0 + 18;   // pre0
    whh_i[1] = b0 + 19; wih_i[1] = b0 + 20; b_i[1] = b0 + 21;   // pre1
    pwhh_i[0] = b0 + 22; pwih_i[0] = b0 + 23; pb_i[0] = b0 + 24;
    pwhh_i[1] = b0 + 25; pwih_i[1] = b0 + 26; pb_i[1] = b0 + 27;
  };
  if (in_sizes[0] == 128 * 64 * 240) {
    xI = 0; xlI = 1;
    if (in_sizes[2] == 4096 * 240) {  // insertion order everywhere
      wih_i[0] = 2;  whh_i[0] = 3;  b_i[0] = 4;
      wih_i[1] = 5;  whh_i[1] = 6;  b_i[1] = 7;
      wih_i[2] = 8;  whh_i[2] = 9;  b_i[2] = 10;
      wih_i[3] = 11; whh_i[3] = 12; b_i[3] = 13;
      wih_i[4] = 14; whh_i[4] = 15; b_i[4] = 16;
      pwih_i[0] = 17; pwhh_i[0] = 18; pb_i[0] = 19;
      pwih_i[1] = 20; pwhh_i[1] = 21; pb_i[1] = 22;
      emb_i = 23; We_i = 24; be_i = 25; Wp_i = 26; bp_i = 27; Wo_i = 28; bo_i = 29;
    } else {
      schemeA(2);
    }
  } else {  // fully sorted top level: params leaves first, then x, x_lens
    schemeA(0);
    xI = 28; xlI = 29;
  }
  auto F = [&](int i) { return (const float*)d_in[i]; };
  const int* x_lens = (const int*)d_in[xlI];
  const float* x = F(xI);

  // ---- workspace layout ----
  size_t off = 0;
  auto A = [&](size_t bytes) -> char* {
    char* p = (char*)d_ws + off;
    off = (off + bytes + 255) & ~(size_t)255;
    return p;
  };
  typedef unsigned short u16;
  // bf16 weights (L2-resident working set, ~90 MB)
  u16* wih[5]; u16* whh[5];
  const int KaL[5] = {256, 1024, 2048, 1024, 1024};   // padded input dims
  const int TL[5]  = {128, 128, 64, 64, 64};
  for (int l = 0; l < 5; ++l) {
    wih[l] = (u16*)A((size_t)4096 * KaL[l] * 2);
    whh[l] = (u16*)A((size_t)4096 * 1024 * 2);
  }
  u16* wpred[2]; wpred[0] = (u16*)A((size_t)1280 * 640 * 2);
  wpred[1] = (u16*)A((size_t)1280 * 640 * 2);
  u16* wWe = (u16*)A((size_t)512 * 1024 * 2);
  u16* wWp = (u16*)A((size_t)512 * 320 * 2);
  u16* wWo = (u16*)A((size_t)32 * 512 * 2);
  u16* embb = (u16*)A((size_t)28 * 320 * 2);
  // sequence buffers (bf16, ping-pong)
  u16* SA = (u16*)A((size_t)64 * 64 * 2048 * 2);  // xbf / stacked / fenc
  u16* SB = (u16*)A((size_t)128 * 64 * 1024 * 2);
  u16* SC = (u16*)A((size_t)128 * 64 * 1024 * 2);
  // recurrent state
  float* zbuf = (float*)A((size_t)64 * 4096 * 4);
  float* hstate = (float*)A((size_t)64 * 1024 * 4);
  float* cstate = (float*)A((size_t)64 * 1024 * 4);
  u16* hbuf = (u16*)A((size_t)64 * 1024 * 2);
  // decode state
  float* hg = (float*)A((size_t)2 * 64 * 320 * 4);
  float* cg = (float*)A((size_t)2 * 64 * 320 * 4);
  float* hgn = (float*)A((size_t)2 * 64 * 320 * 4);
  float* cgn = (float*)A((size_t)2 * 64 * 320 * 4);
  float* zp = (float*)A((size_t)64 * 1280 * 4);
  float* zj = (float*)A((size_t)64 * 512 * 4);
  float* logits = (float*)A((size_t)64 * 32 * 4);
  float* bj = (float*)A(512 * 4);
  float* bo32 = (float*)A(32 * 4);
  u16* act = (u16*)A((size_t)64 * 640 * 2);
  u16* gout = (u16*)A((size_t)64 * 320 * 2);
  u16* hj = (u16*)A((size_t)64 * 512 * 2);
  int* iscr = (int*)A(16 * 64 * 4);
  int *pre_g = iscr, *sym_add = iscr + 64, *tidv = iscr + 128, *finishv = iscr + 192,
      *res_idx = iscr + 256, *flen = iscr + 320, *eosv = iscr + 384, *upd = iscr + 448,
      *fi_rows = iscr + 512, *done = iscr + 576;

  // ---- weight / input conversion to bf16 (per call; deterministic) ----
  auto cvt = [&](u16* dst, int dstLd, int colOff, const float* src, int srcRows,
                 int srcCols, int padRows, int padCols) {
    long total = (long)padRows * padCols;
    cvt_bf16_kernel<<<(unsigned)CDIV(total, 256), 256, 0, stream>>>(
        dst, dstLd, colOff, src, srcRows, srcCols, padRows, padCols);
  };
  const int inDim[5] = {240, 1024, 2048, 1024, 1024};
  for (int l = 0; l < 5; ++l) {
    cvt(wih[l], KaL[l], 0, F(wih_i[l]), 4096, inDim[l], 4096, KaL[l]);
    cvt(whh[l], 1024, 0, F(whh_i[l]), 4096, 1024, 4096, 1024);
  }
  for (int l = 0; l < 2; ++l) {
    cvt(wpred[l], 640, 0,   F(pwih_i[l]), 1280, 320, 1280, 320);
    cvt(wpred[l], 640, 320, F(pwhh_i[l]), 1280, 320, 1280, 320);
  }
  cvt(wWe, 1024, 0, F(We_i), 512, 1024, 512, 1024);
  cvt(wWp, 320, 0, F(Wp_i), 512, 320, 512, 320);
  cvt(wWo, 512, 0, F(Wo_i), 29, 512, 32, 512);
  cvt(embb, 320, 0, F(emb_i), 28, 320, 28, 320);
  cvt(SA, 256, 0, x, 128 * 64, 240, 128 * 64, 256);  // xbf (padded 240->256)

  // ---- transcription: 2 pre layers, time-stack, 3 post layers ----
  auto run_layer = [&](const u16* seqIn, int l, u16* seqOut) {
    zero_state_kernel<<<CDIV(64 * 1024, 256), 256, 0, stream>>>(hstate, cstate, hbuf,
                                                                64 * 1024);
    int Ka = KaL[l];
    for (int t = 0; t < TL[l]; ++t) {
      lstm_gates_kernel<<<32, 256, 0, stream>>>(
          seqIn + (size_t)t * 64 * Ka, Ka, Ka, wih[l], hbuf, 1024, whh[l],
          F(b_i[l]), zbuf, 4096);
      lstm_pointwise_kernel<<<CDIV(64 * 1024, 256), 256, 0, stream>>>(
          zbuf, hstate, cstate, hbuf, seqOut + (size_t)t * 64 * 1024);
    }
  };
  run_layer(SA, 0, SB);                       // pre0: xbf -> SB
  run_layer(SB, 1, SC);                       // pre1: SB -> SC
  stack_kernel<<<CDIV(64 * 64 * 2048, 256), 256, 0, stream>>>(SA, SC);  // SC -> SA
  run_layer(SA, 2, SB);                       // post0: stacked -> SB
  run_layer(SB, 3, SC);                       // post1
  run_layer(SC, 4, SA);                       // post2 -> fenc in SA

  // ---- greedy decode (persistent single-workgroup kernel) ----
  int* res = (int*)d_out;
  int* counts = res + 64 * 3840;
  decode_init_kernel<<<CDIV(64 * 3840, 256), 256, 0, stream>>>(
      res, pre_g, sym_add, tidv, finishv, res_idx, flen, eosv, upd, fi_rows, done,
      hg, cg, bj, F(be_i), F(bp_i), bo32, F(bo_i), x_lens);

  DecodeArgs da;
  da.fenc = SA; da.wpred0 = wpred[0]; da.wpred1 = wpred[1];
  da.wWe = wWe; da.wWp = wWp; da.wWo = wWo; da.embb = embb;
  da.bpred0 = F(pb_i[0]); da.bpred1 = F(pb_i[1]); da.bj = bj; da.bo32 = bo32;
  da.hg = hg; da.cg = cg; da.hgn = hgn; da.cgn = cgn;
  da.zp = zp; da.zj = zj; da.logits = logits;
  da.act = act; da.gout = gout; da.hj = hj;
  da.pre_g = pre_g; da.sym_add = sym_add; da.tidv = tidv; da.finishv = finishv;
  da.res_idx = res_idx; da.flen = flen; da.eosv = eosv; da.upd = upd;
  da.fi_rows = fi_rows; da.done = done; da.res = res; da.counts = counts;
  decode_kernel<<<1, 1024, 0, stream>>>(da);
}
